// Global_Critic_63247688401368
// MI455X (gfx1250) — matrix operand order repaired and checked by reading, not compiled, not run on hardware
//
#include <hip/hip_runtime.h>
#include <hip/hip_bf16.h>
#include <stddef.h>
#include <stdint.h>

// ---------------- problem constants ----------------
constexpr int A_ = 8;       // agents
constexpr int B_ = 32768;   // batch
constexpr int O_ = 256;     // obs features
constexpr int C_ = 64;      // action features
constexpr int E_ = 128;     // embedding
constexpr int F_ = 320;     // O_ + C_
constexpr float EPS_   = 1e-5f;
constexpr float SLOPE_ = 0.01f;

// LDS strides (f16 units), chosen 16B-aligned
constexpr int XS = 40;      // activation chunk row stride (32 used)
constexpr int WS = 40;      // weight chunk row stride (32 used)
constexpr int ES = 136;     // e/h tile row stride (128 used)
constexpr int QS = 68;      // q tile row stride (f32)

typedef _Float16 half_t;
typedef __attribute__((ext_vector_type(16))) _Float16 v16h;
typedef __attribute__((ext_vector_type(8)))  _Float16 v8h;
typedef __attribute__((ext_vector_type(8)))  float    v8f;
typedef __attribute__((ext_vector_type(4)))  float    v4f;
typedef int b128_t __attribute__((vector_size(16)));   // matches builtin param type

// --------- gfx1250 async global->LDS copy (ASYNCcnt path) ---------
#if defined(__AMDGCN__) &&                                              \
    __has_builtin(__builtin_amdgcn_global_load_async_to_lds_b128) &&    \
    __has_builtin(__builtin_amdgcn_s_wait_asynccnt)
#define HAVE_ASYNC_LDS 1
#else
#define HAVE_ASYNC_LDS 0
#endif

// 16-byte global->LDS copy. Uses GLOBAL_LOAD_ASYNC_TO_LDS_B128 when the
// toolchain exposes it; falls back to load+ds_store otherwise.
// AS conversion via inttoptr: ISA 10.2 — generic-LDS addr low 32 bits ARE the
// LDS byte address, so truncation yields a valid AS(3) pointer.
__device__ __forceinline__ void async_copy_b128(const half_t* gsrc, half_t* ldst) {
#if HAVE_ASYNC_LDS
  __builtin_amdgcn_global_load_async_to_lds_b128(
      (__attribute__((address_space(1))) b128_t*)(uintptr_t)gsrc,
      (__attribute__((address_space(3))) b128_t*)(uint32_t)(uintptr_t)ldst,
      /*offset=*/0, /*cpol=*/0);
#else
  *(v8h*)ldst = *(const v8h*)gsrc;
#endif
}

__device__ __forceinline__ void wait_async() {
#if HAVE_ASYNC_LDS
  __builtin_amdgcn_s_wait_asynccnt(0);
#endif
}

__device__ __forceinline__ float lrelu(float x) {
  return x >= 0.0f ? x : SLOPE_ * x;
}

// D = A(16x32 f16) x B(32x16 f16) + C(16x16 f32)
__device__ __forceinline__ v8f wmma_f16(v16h a, v16h b, v8f c) {
  return __builtin_amdgcn_wmma_f32_16x16x32_f16(
      /*neg_a=*/false, a, /*neg_b=*/false, b,
      /*c_mod=*/(short)0, c, /*reuse_a=*/false, /*reuse_b=*/false);
}

// A fragment: lane holds row m; a[j]=A[m][hf*8+j] (j<8), a[8+j]=A[m][16+hf*8+j]
__device__ __forceinline__ v16h load_frag_a(const half_t* row, int hf) {
  union { v16h v; v8h h[2]; } u;
  u.h[0] = *(const v8h*)(row + hf * 8);
  u.h[1] = *(const v8h*)(row + 16 + hf * 8);
  return u.v;
}
// B fragment: B[k][n] = W[n][k]; lane holds col n; b[j]=W[n][hf*8+j] (j<8), b[8+j]=W[n][16+hf*8+j]
__device__ __forceinline__ v16h load_frag_b(const half_t* wrow, int hf) {
  union { v16h v; v8h h[2]; } u;
  u.h[0] = *(const v8h*)(wrow + hf * 8);
  u.h[1] = *(const v8h*)(wrow + 16 + hf * 8);
  return u.v;
}

// ---------------- small prep kernels ----------------
__global__ void k_zero(float* p, int n) {
  int i = blockIdx.x * blockDim.x + threadIdx.x;
  if (i < n) p[i] = 0.0f;
}

// per-(agent,feature) sum / sumsq over batch (combined = [obs, act])
__global__ __launch_bounds__(256) void k_stats(
    const float* __restrict__ obs, const float* __restrict__ act,
    float* __restrict__ sum, float* __restrict__ sumsq) {
  const int a  = blockIdx.x >> 7;        // /128
  const int b0 = (blockIdx.x & 127) * 256;
  const int t  = threadIdx.x;
  float s = 0.f, s2 = 0.f, sa = 0.f, sa2 = 0.f;
  for (int i = 0; i < 256; ++i) {
    float x = obs[((size_t)(a * B_ + b0 + i)) * O_ + t];
    s += x; s2 += x * x;
    if (t < C_) {
      float y = act[((size_t)(a * B_ + b0 + i)) * C_ + t];
      sa += y; sa2 += y * y;
    }
  }
  atomicAdd(&sum[a * F_ + t], s);
  atomicAdd(&sumsq[a * F_ + t], s2);
  if (t < C_) {
    atomicAdd(&sum[a * F_ + O_ + t], sa);
    atomicAdd(&sumsq[a * F_ + O_ + t], sa2);
  }
}

// BN fold factors: xn = x*scale + shift
__global__ void k_bnprep(const float* __restrict__ sum, const float* __restrict__ sumsq,
                         const float* __restrict__ gG, const float* __restrict__ gB,
                         const float* __restrict__ sG, const float* __restrict__ sB,
                         float* __restrict__ scale_g, float* __restrict__ shift_g,
                         float* __restrict__ scale_s, float* __restrict__ shift_s) {
  const int a = blockIdx.x, f = threadIdx.x;  // 320 threads
  float mu  = sum[a * F_ + f] * (1.0f / B_);
  float var = sumsq[a * F_ + f] * (1.0f / B_) - mu * mu;
  float inv = rsqrtf(var + EPS_);
  float sg = gG[a * F_ + f] * inv;
  scale_g[a * F_ + f] = sg;
  shift_g[a * F_ + f] = gB[a * F_ + f] - mu * sg;
  if (f < O_) {
    float ss = sG[a * O_ + f] * inv;
    scale_s[a * O_ + f] = ss;
    shift_s[a * O_ + f] = sB[a * O_ + f] - mu * ss;
  }
}

// fold BN into g_W / s_W (-> f16) and biases
__global__ __launch_bounds__(64) void k_fold(
    const float* __restrict__ gW, const float* __restrict__ gb,
    const float* __restrict__ sW, const float* __restrict__ sb,
    const float* __restrict__ scale_g, const float* __restrict__ shift_g,
    const float* __restrict__ scale_s, const float* __restrict__ shift_s,
    half_t* __restrict__ gWh, float* __restrict__ gbh,
    half_t* __restrict__ sWh, float* __restrict__ sbh) {
  const int a = blockIdx.x >> 7, e = blockIdx.x & 127, t = threadIdx.x;
  __shared__ float redg[64], reds[64];
  float ag = 0.f, as = 0.f;
  for (int f = t; f < F_; f += 64) {
    float w = gW[((size_t)(a * E_ + e)) * F_ + f];
    gWh[((size_t)(a * E_ + e)) * F_ + f] = (half_t)(w * scale_g[a * F_ + f]);
    ag += w * shift_g[a * F_ + f];
  }
  for (int f = t; f < O_; f += 64) {
    float w = sW[((size_t)(a * E_ + e)) * O_ + f];
    sWh[((size_t)(a * E_ + e)) * O_ + f] = (half_t)(w * scale_s[a * O_ + f]);
    as += w * shift_s[a * O_ + f];
  }
  redg[t] = ag; reds[t] = as;
  __syncthreads();
  for (int s = 32; s > 0; s >>= 1) {
    if (t < s) { redg[t] += redg[t + s]; reds[t] += reds[t + s]; }
    __syncthreads();
  }
  if (t == 0) {
    gbh[a * E_ + e] = gb[a * E_ + e] + redg[0];
    sbh[a * E_ + e] = sb[a * E_ + e] + reds[0];
  }
}

// convert V / f_W1 / f_W2 to f16
__global__ void k_cvt(const float* __restrict__ V, const float* __restrict__ fW1,
                      const float* __restrict__ fW2, half_t* __restrict__ Vh,
                      half_t* __restrict__ fW1h, half_t* __restrict__ fW2h) {
  const int NV = E_ * E_, N1 = A_ * E_ * 2 * E_, N2 = A_ * C_ * E_;
  int i = blockIdx.x * blockDim.x + threadIdx.x;
  if (i < NV) Vh[i] = (half_t)V[i];
  else if (i < NV + N1) fW1h[i - NV] = (half_t)fW1[i - NV];
  else if (i < NV + N1 + N2) fW2h[i - NV - N1] = (half_t)fW2[i - NV - N1];
}

__global__ void k_argmax(const float* __restrict__ act, int* __restrict__ ids) {
  int idx = blockIdx.x * blockDim.x + threadIdx.x;   // over A*B
  const float* r = act + (size_t)idx * C_;
  float best = r[0]; int bi = 0;
  for (int j = 1; j < C_; ++j) { float v = r[j]; if (v > best) { best = v; bi = j; } }
  ids[idx] = bi;
}

// ---------------- stage 1: e, si, v ----------------
__global__ __launch_bounds__(256) void k_stage1(
    const float* __restrict__ obs, const float* __restrict__ act,
    const half_t* __restrict__ gWh, const float* __restrict__ gbh,
    const half_t* __restrict__ sWh, const float* __restrict__ sbh,
    const half_t* __restrict__ Vh,
    half_t* __restrict__ v_out, half_t* __restrict__ si_out) {
  __shared__ half_t sX[64 * XS];
  __shared__ half_t sWg[128 * WS];
  __shared__ half_t sWs[128 * WS];
  __shared__ half_t sE[64 * ES];

  const int tid = threadIdx.x;
  const int a  = blockIdx.x >> 9;
  const int b0 = (blockIdx.x & 511) * 64;
  const int wave = tid >> 5, lane = tid & 31;
  const int hf = lane >> 4, lr = lane & 15;
  const int mblk = wave & 3, nbase = (wave >> 2) * 4;
  const int m = mblk * 16 + lr;
  const int srow = tid >> 2, sc0 = (tid & 3) * 8;   // activation staging
  const int wrow = tid >> 1, wc0 = (tid & 1) * 16;  // weight staging

  v8f accE[4], accS[4];
  v8f zero = {0.f, 0.f, 0.f, 0.f, 0.f, 0.f, 0.f, 0.f};
#pragma unroll
  for (int j = 0; j < 4; ++j) { accE[j] = zero; accS[j] = zero; }

  // --- GEMM e (K=320) fused with GEMM si (K=256, shares obs staging) ---
  for (int kc = 0; kc < 10; ++kc) {
    // weight chunks: pure byte copies -> async global->LDS (ASYNCcnt)
    {
      const half_t* src = gWh + ((size_t)(a * E_ + wrow)) * F_ + kc * 32 + wc0;
      async_copy_b128(src,     &sWg[wrow * WS + wc0]);
      async_copy_b128(src + 8, &sWg[wrow * WS + wc0 + 8]);
    }
    if (kc < 8) {
      const half_t* src = sWh + ((size_t)(a * E_ + wrow)) * O_ + kc * 32 + wc0;
      async_copy_b128(src,     &sWs[wrow * WS + wc0]);
      async_copy_b128(src + 8, &sWs[wrow * WS + wc0 + 8]);
    }
    // activation chunk: f32 -> f16 conversion, VGPR path
    {
      const float* src = (kc < 8)
          ? obs + ((size_t)(a * B_ + b0 + srow)) * O_ + kc * 32 + sc0
          : act + ((size_t)(a * B_ + b0 + srow)) * C_ + (kc - 8) * 32 + sc0;
      v4f x0 = *(const v4f*)src;
      v4f x1 = *(const v4f*)(src + 4);
      half_t* d = &sX[srow * XS + sc0];
      d[0] = (half_t)x0[0]; d[1] = (half_t)x0[1]; d[2] = (half_t)x0[2]; d[3] = (half_t)x0[3];
      d[4] = (half_t)x1[0]; d[5] = (half_t)x1[1]; d[6] = (half_t)x1[2]; d[7] = (half_t)x1[3];
    }
    wait_async();
    __syncthreads();
    v16h af = load_frag_a(&sX[m * XS], hf);
#pragma unroll
    for (int j = 0; j < 4; ++j) {
      int n = (nbase + j) * 16 + lr;
      v16h bg = load_frag_b(&sWg[n * WS], hf);
      accE[j] = wmma_f16(af, bg, accE[j]);
      if (kc < 8) {
        v16h bs = load_frag_b(&sWs[n * WS], hf);
        accS[j] = wmma_f16(af, bs, accS[j]);
      }
    }
    __syncthreads();
  }

  // epilogue: bias + lrelu; e -> LDS (f16), si -> global (f16)
#pragma unroll
  for (int j = 0; j < 4; ++j) {
    int n = (nbase + j) * 16 + lr;
    float bg = gbh[a * E_ + n];
    float bs = sbh[a * E_ + n];
#pragma unroll
    for (int r = 0; r < 8; ++r) {
      int row = mblk * 16 + hf * 8 + r;
      sE[row * ES + n] = (half_t)lrelu(accE[j][r] + bg);
      si_out[((size_t)(a * B_ + b0 + row)) * E_ + n] = (half_t)lrelu(accS[j][r] + bs);
    }
  }
  __syncthreads();

  // --- GEMM v = lrelu(e @ V^T), K=128 ---
  v8f accV[4];
#pragma unroll
  for (int j = 0; j < 4; ++j) accV[j] = zero;
  for (int kc = 0; kc < 4; ++kc) {
    {
      const half_t* src = Vh + (size_t)wrow * E_ + kc * 32 + wc0;
      async_copy_b128(src,     &sWg[wrow * WS + wc0]);
      async_copy_b128(src + 8, &sWg[wrow * WS + wc0 + 8]);
    }
    wait_async();
    __syncthreads();
    v16h af = load_frag_a(&sE[m * ES + kc * 32], hf);
#pragma unroll
    for (int j = 0; j < 4; ++j) {
      int n = (nbase + j) * 16 + lr;
      v16h bv = load_frag_b(&sWg[n * WS], hf);
      accV[j] = wmma_f16(af, bv, accV[j]);
    }
    __syncthreads();
  }
#pragma unroll
  for (int j = 0; j < 4; ++j) {
    int n = (nbase + j) * 16 + lr;
#pragma unroll
    for (int r = 0; r < 8; ++r) {
      int row = mblk * 16 + hf * 8 + r;
      v_out[((size_t)(a * B_ + b0 + row)) * E_ + n] = (half_t)lrelu(accV[j][r]);
    }
  }
}

// vsum[b][e] = sum_a v[a][b][e]
__global__ void k_vsum(const half_t* __restrict__ v, float* __restrict__ vsum) {
  int idx = blockIdx.x * blockDim.x + threadIdx.x;  // over B_*E_
  float s = 0.f;
#pragma unroll
  for (int a = 0; a < A_; ++a) s += (float)v[(size_t)a * B_ * E_ + idx];
  vsum[idx] = s;
}

// ---------------- stage 2: h, q, gather ----------------
__global__ __launch_bounds__(256) void k_stage2(
    const half_t* __restrict__ v_in, const float* __restrict__ vsum,
    const half_t* __restrict__ si_in,
    const half_t* __restrict__ fW1h, const float* __restrict__ fb1,
    const half_t* __restrict__ fW2h, const float* __restrict__ fb2,
    const int* __restrict__ ids, float* __restrict__ out) {
  __shared__ half_t sX[64 * XS];
  __shared__ half_t sW[128 * WS];
  __shared__ half_t sE[64 * ES];
  __shared__ float  sQ[64 * QS];

  const int tid = threadIdx.x;
  const int a  = blockIdx.x >> 9;
  const int b0 = (blockIdx.x & 511) * 64;
  const int wave = tid >> 5, lane = tid & 31;
  const int hf = lane >> 4, lr = lane & 15;
  const int mblk = wave & 3, nbase = (wave >> 2) * 4;
  const int m = mblk * 16 + lr;
  const int srow = tid >> 2, sc0 = (tid & 3) * 8;
  const int wrow = tid >> 1, wc0 = (tid & 1) * 16;

  v8f zero = {0.f, 0.f, 0.f, 0.f, 0.f, 0.f, 0.f, 0.f};
  v8f accH[4];
#pragma unroll
  for (int j = 0; j < 4; ++j) accH[j] = zero;

  // --- h = lrelu([vsum - v, si] @ f_W1^T + b1), K=256 ---
  for (int kc = 0; kc < 8; ++kc) {
    {
      const half_t* src = fW1h + ((size_t)(a * E_ + wrow)) * (2 * E_) + kc * 32 + wc0;
      async_copy_b128(src,     &sW[wrow * WS + wc0]);
      async_copy_b128(src + 8, &sW[wrow * WS + wc0 + 8]);
    }
    if (kc < 4) {  // x = vsum - v (compute path)
      const float*  vs = vsum + ((size_t)(b0 + srow)) * E_ + kc * 32 + sc0;
      const half_t* vp = v_in + ((size_t)(a * B_ + b0 + srow)) * E_ + kc * 32 + sc0;
      v4f s0 = *(const v4f*)vs;
      v4f s1 = *(const v4f*)(vs + 4);
      v8h vv = *(const v8h*)vp;
      half_t* d = &sX[srow * XS + sc0];
      d[0] = (half_t)(s0[0] - (float)vv[0]); d[1] = (half_t)(s0[1] - (float)vv[1]);
      d[2] = (half_t)(s0[2] - (float)vv[2]); d[3] = (half_t)(s0[3] - (float)vv[3]);
      d[4] = (half_t)(s1[0] - (float)vv[4]); d[5] = (half_t)(s1[1] - (float)vv[5]);
      d[6] = (half_t)(s1[2] - (float)vv[6]); d[7] = (half_t)(s1[3] - (float)vv[7]);
    } else {       // si: already f16 -> async byte copy
      const half_t* sp = si_in + ((size_t)(a * B_ + b0 + srow)) * E_ + (kc - 4) * 32 + sc0;
      async_copy_b128(sp, &sX[srow * XS + sc0]);
    }
    wait_async();
    __syncthreads();
    v16h af = load_frag_a(&sX[m * XS], hf);
#pragma unroll
    for (int j = 0; j < 4; ++j) {
      int n = (nbase + j) * 16 + lr;
      v16h b = load_frag_b(&sW[n * WS], hf);
      accH[j] = wmma_f16(af, b, accH[j]);
    }
    __syncthreads();
  }
#pragma unroll
  for (int j = 0; j < 4; ++j) {
    int n = (nbase + j) * 16 + lr;
    float b1 = fb1[a * E_ + n];
#pragma unroll
    for (int r = 0; r < 8; ++r) {
      int row = mblk * 16 + hf * 8 + r;
      sE[row * ES + n] = (half_t)lrelu(accH[j][r] + b1);
    }
  }
  __syncthreads();

  // --- q = h @ f_W2^T + b2, N=64, K=128 ---
  const int nbase2 = (wave >> 2) * 2;  // waves 0-3 -> tiles {0,1}; 4-7 -> {2,3}
  v8f accQ[2];
  accQ[0] = zero; accQ[1] = zero;
  for (int kc = 0; kc < 4; ++kc) {
    if (tid < 128) {
      int r2 = tid >> 1, c2 = (tid & 1) * 16;
      const half_t* src = fW2h + ((size_t)(a * C_ + r2)) * E_ + kc * 32 + c2;
      async_copy_b128(src,     &sW[r2 * WS + c2]);
      async_copy_b128(src + 8, &sW[r2 * WS + c2 + 8]);
    }
    wait_async();
    __syncthreads();
    v16h af = load_frag_a(&sE[m * ES + kc * 32], hf);
#pragma unroll
    for (int j = 0; j < 2; ++j) {
      int n = (nbase2 + j) * 16 + lr;
      v16h b = load_frag_b(&sW[n * WS], hf);
      accQ[j] = wmma_f16(af, b, accQ[j]);
    }
    __syncthreads();
  }
#pragma unroll
  for (int j = 0; j < 2; ++j) {
    int n = (nbase2 + j) * 16 + lr;
    float b2 = fb2[a * C_ + n];
#pragma unroll
    for (int r = 0; r < 8; ++r) {
      int row = mblk * 16 + hf * 8 + r;
      sQ[row * QS + n] = accQ[j][r] + b2;
    }
  }
  __syncthreads();

  // gather argmax(action) column
  if (tid < 64) {
    int id = ids[(size_t)a * B_ + b0 + tid];
    out[(size_t)a * B_ + b0 + tid] = sQ[tid * QS + id];
  }
}

// ---------------- host launch ----------------
extern "C" void kernel_launch(void* const* d_in, const int* in_sizes, int n_in,
                              void* d_out, int out_size, void* d_ws, size_t ws_size,
                              hipStream_t stream) {
  const float* obs  = (const float*)d_in[0];
  const float* act  = (const float*)d_in[1];
  const float* gG   = (const float*)d_in[2];
  const float* gB   = (const float*)d_in[3];
  const float* gW   = (const float*)d_in[4];
  const float* gb   = (const float*)d_in[5];
  const float* sG   = (const float*)d_in[6];
  const float* sB   = (const float*)d_in[7];
  const float* sW   = (const float*)d_in[8];
  const float* sb   = (const float*)d_in[9];
  const float* fW1  = (const float*)d_in[10];
  const float* fb1  = (const float*)d_in[11];
  const float* fW2  = (const float*)d_in[12];
  const float* fb2  = (const float*)d_in[13];
  // d_in[14]=Wq, d_in[15]=Wk unused (softmax over size-1 axis == 1)
  const float* V    = (const float*)d_in[16];
  float* out = (float*)d_out;

  char* w = (char*)d_ws;
  auto alloc = [&](size_t bytes) -> void* {
    void* p = (void*)w;
    w += (bytes + 255) & ~(size_t)255;
    return p;
  };
  float*  ws_sum   = (float*)alloc(A_ * F_ * 4);           // followed contiguously by sumsq
  float*  ws_sumsq = (float*)alloc(A_ * F_ * 4);
  float*  ws_scg   = (float*)alloc(A_ * F_ * 4);
  float*  ws_shg   = (float*)alloc(A_ * F_ * 4);
  float*  ws_scs   = (float*)alloc(A_ * O_ * 4);
  float*  ws_shs   = (float*)alloc(A_ * O_ * 4);
  half_t* ws_gWh   = (half_t*)alloc((size_t)A_ * E_ * F_ * 2);
  float*  ws_gbh   = (float*)alloc(A_ * E_ * 4);
  half_t* ws_sWh   = (half_t*)alloc((size_t)A_ * E_ * O_ * 2);
  float*  ws_sbh   = (float*)alloc(A_ * E_ * 4);
  half_t* ws_Vh    = (half_t*)alloc((size_t)E_ * E_ * 2);
  half_t* ws_fW1h  = (half_t*)alloc((size_t)A_ * E_ * 2 * E_ * 2);
  half_t* ws_fW2h  = (half_t*)alloc((size_t)A_ * C_ * E_ * 2);
  half_t* ws_v     = (half_t*)alloc((size_t)A_ * B_ * E_ * 2);
  float*  ws_vsum  = (float*)alloc((size_t)B_ * E_ * 4);
  half_t* ws_si    = (half_t*)alloc((size_t)A_ * B_ * E_ * 2);
  int*    ws_ids   = (int*)alloc((size_t)A_ * B_ * 4);

  // 1) BN statistics
  k_zero<<<(2 * A_ * F_ + 255) / 256, 256, 0, stream>>>(ws_sum, 2 * A_ * F_);
  k_stats<<<A_ * 128, 256, 0, stream>>>(obs, act, ws_sum, ws_sumsq);
  // 2) fold BN into weights, convert to f16
  k_bnprep<<<A_, F_, 0, stream>>>(ws_sum, ws_sumsq, gG, gB, sG, sB,
                                  ws_scg, ws_shg, ws_scs, ws_shs);
  k_fold<<<A_ * E_, 64, 0, stream>>>(gW, gb, sW, sb, ws_scg, ws_shg, ws_scs, ws_shs,
                                     ws_gWh, ws_gbh, ws_sWh, ws_sbh);
  {
    int ntot = E_ * E_ + A_ * E_ * 2 * E_ + A_ * C_ * E_;
    k_cvt<<<(ntot + 255) / 256, 256, 0, stream>>>(V, fW1, fW2, ws_Vh, ws_fW1h, ws_fW2h);
  }
  // 3) argmax indices
  k_argmax<<<(A_ * B_) / 256, 256, 0, stream>>>(act, ws_ids);
  // 4) stage 1: e, si, v (WMMA)
  k_stage1<<<A_ * (B_ / 64), 256, 0, stream>>>(obs, act, ws_gWh, ws_gbh, ws_sWh, ws_sbh,
                                               ws_Vh, ws_v, ws_si);
  // 5) cross-agent sum
  k_vsum<<<(B_ * E_) / 256, 256, 0, stream>>>(ws_v, ws_vsum);
  // 6) stage 2: h, q, gather (WMMA)
  k_stage2<<<A_ * (B_ / 64), 256, 0, stream>>>(ws_v, ws_vsum, ws_si, ws_fW1h, fb1,
                                               ws_fW2h, fb2, ws_ids, out);
  (void)in_sizes; (void)n_in; (void)out_size; (void)ws_size;
}
